// MMDLoss_8907762171929
// MI455X (gfx1250) — compile-verified, matrix operand order checked by reading, not run on hardware
//
#include <hip/hip_runtime.h>

// ---------------------------------------------------------------------------
// MMD loss (Gaussian kernel, sigma=1) on gfx1250 via bf16 split-WMMA with
// async global->LDS DMA staging (ASYNCcnt), double-buffered LDS, 2x4 wave
// tiling, and symmetric-triangle pruning for the XX / YY Gram matrices.
//   dot(x,y) ~= hi_x.hi_y + hi_x.lo_y + lo_x.hi_y   (f32 accumulate in WMMA)
//   dist     = |x|^2 + |y|^2 - 2 dot
//   mmd      = mean(exp(-dist/2)) over XX + YY - 2*XY
// ---------------------------------------------------------------------------

typedef __attribute__((ext_vector_type(16))) __bf16        v16bf;
typedef __attribute__((ext_vector_type(8)))  float         v8f;
typedef __attribute__((ext_vector_type(4)))  unsigned int  u32x4;
typedef int vec4i __attribute__((vector_size(16)));

#define AS1 __attribute__((address_space(1)))
#define AS3 __attribute__((address_space(3)))

#define D_DIM   512  // feature dimension (fixed by reference)
#define TILE    128  // output tile per workgroup (128x128)
#define KC       32  // K-chunk per WMMA step
#define STRIDE   40  // LDS row stride in bf16 (80 B: 16B-aligned, conflict-free)

#define ARR_BYTES  (TILE * STRIDE * 2)   // 10240 B per staging array
#define BUF_BYTES  (4 * ARR_BYTES)       // Ah,Al,Bh,Bl per buffer = 40960 B
#define SMEM_BYTES (2 * BUF_BYTES + 256 * 4)

// Load one 16x32 bf16 WMMA operand fragment from a 16-row LDS tile.
// ISA 7.12.2 (16-bit A/B layout): lane L -> row M = L%16, K-chunks at
// (L/16)*8 .. +7 and (L/16)*8+16 .. +23  => two aligned 16-byte LDS reads.
__device__ __forceinline__ v16bf load_frag(const __bf16* base, int lane) {
  const int rl   = lane & 15;
  const int koff = (lane >> 4) * 8;
  const __bf16* p = base + rl * STRIDE + koff;
  union { v16bf v; u32x4 q[2]; } f;
  f.q[0] = *(const u32x4*)(p);
  f.q[1] = *(const u32x4*)(p + 16);
  return f.v;
}

// Issue async global->LDS DMA for one 128x32 hi/lo A/B slab (8 b128 per lane).
__device__ __forceinline__ void async_stage(
    const __bf16* __restrict__ Ah_g, const __bf16* __restrict__ Al_g,
    const __bf16* __restrict__ Bh_g, const __bf16* __restrict__ Bl_g,
    char* smembase, int buf, int tid, int rowBase, int colBase, int kc) {
  char* lbase = smembase + buf * BUF_BYTES;
  #pragma unroll
  for (int it = 0; it < 2; ++it) {
    const int i = tid + it * 256;
    const int r = i >> 2;
    const int c = (i & 3) * 8;
    const size_t gA = (size_t)(rowBase + r) * D_DIM + kc + c;
    const size_t gB = (size_t)(colBase + r) * D_DIM + kc + c;
    char* l = lbase + r * (STRIDE * 2) + c * 2;
    __builtin_amdgcn_global_load_async_to_lds_b128(
        (AS1 vec4i*)(Ah_g + gA), (AS3 vec4i*)(l + 0 * ARR_BYTES), 0, 0);
    __builtin_amdgcn_global_load_async_to_lds_b128(
        (AS1 vec4i*)(Al_g + gA), (AS3 vec4i*)(l + 1 * ARR_BYTES), 0, 0);
    __builtin_amdgcn_global_load_async_to_lds_b128(
        (AS1 vec4i*)(Bh_g + gB), (AS3 vec4i*)(l + 2 * ARR_BYTES), 0, 0);
    __builtin_amdgcn_global_load_async_to_lds_b128(
        (AS1 vec4i*)(Bl_g + gB), (AS3 vec4i*)(l + 3 * ARR_BYTES), 0, 0);
  }
}

// One wave (32 lanes) per row: produce bf16 hi/lo split and the f32 row norm
// computed with the *same* arithmetic the WMMA path uses (hh + 2*hl), so the
// Gram diagonal cancels to ~0 in the distance computation.
__global__ __launch_bounds__(256) void convert_rows(
    const float* __restrict__ x, __bf16* __restrict__ hi,
    __bf16* __restrict__ lo, float* __restrict__ norms, int N) {
  const int wave = threadIdx.x >> 5, lane = threadIdx.x & 31;
  const int row  = blockIdx.x * 8 + wave;
  if (row >= N) return;
  const float* xr = x  + (size_t)row * D_DIM;
  __bf16*      hr = hi + (size_t)row * D_DIM;
  __bf16*      lr = lo + (size_t)row * D_DIM;
  float s = 0.f;
  #pragma unroll 4
  for (int k = lane; k < D_DIM; k += 32) {
    const float v  = xr[k];
    const __bf16 h = (__bf16)v;
    const float hf = (float)h;
    const __bf16 l = (__bf16)(v - hf);
    const float lf = (float)l;
    hr[k] = h;
    lr[k] = l;
    s += hf * hf + 2.f * hf * lf;
  }
  #pragma unroll
  for (int off = 16; off >= 1; off >>= 1) s += __shfl_xor(s, off);
  if (lane == 0) norms[row] = s;
}

__global__ void zero_out_kernel(float* out) { out[0] = 0.f; }

__global__ __launch_bounds__(256) void mmd_tile_kernel(
    const __bf16* __restrict__ sHi, const __bf16* __restrict__ sLo,
    const __bf16* __restrict__ tHi, const __bf16* __restrict__ tLo,
    const float* __restrict__ sNorm, const float* __restrict__ tNorm,
    float* __restrict__ out, int N, int M) {
  extern __shared__ __align__(16) char smem[];  // 2 x (Ah,Al,Bh,Bl) + red[]

  // z: 0 = XX (+1), 1 = YY (+1), 2 = XY (-2).
  // XX / YY are symmetric: skip blocks strictly below the diagonal and count
  // strictly-upper blocks twice (exact — K(i,j) == K(j,i)).
  const int z = blockIdx.z;
  if (z < 2 && blockIdx.x > blockIdx.y) return;

  const int tid  = threadIdx.x;
  const int wave = tid >> 5;
  const int lane = tid & 31;

  const __bf16 *Ah_g, *Al_g, *Bh_g, *Bl_g;
  const float *nA, *nB;
  float w; int dA, dB;
  const float sym = (blockIdx.x == blockIdx.y) ? 1.f : 2.f;
  if (z == 0)      { Ah_g = sHi; Al_g = sLo; Bh_g = sHi; Bl_g = sLo; nA = sNorm; nB = sNorm; w = sym;  dA = N; dB = N; }
  else if (z == 1) { Ah_g = tHi; Al_g = tLo; Bh_g = tHi; Bl_g = tLo; nA = tNorm; nB = tNorm; w = sym;  dA = M; dB = M; }
  else             { Ah_g = sHi; Al_g = sLo; Bh_g = tHi; Bl_g = tLo; nA = sNorm; nB = tNorm; w = -2.f; dA = N; dB = M; }

  const int rowBase = blockIdx.x * TILE;
  const int colBase = blockIdx.y * TILE;

  // 2x4 wave tiling: wave w -> 32-row x 64-col patch (rg = w>>1, cg = w&1),
  // i.e. 2 row tiles x 4 col tiles of 16x16. Cuts per-wave LDS fragment
  // loads from 18 to 12 per K-chunk vs a 1x8 arrangement.
  const int rg = wave >> 1;            // 0..3  -> rows rg*32 .. +31
  const int cg = wave & 1;             // 0..1  -> cols cg*64 .. +63

  v8f acc[8] = {};                     // acc[i*4+j]: row tile i, col tile j

  // Prologue: DMA chunk 0 into buffer 0.
  async_stage(Ah_g, Al_g, Bh_g, Bl_g, smem, 0, tid, rowBase, colBase, 0);

  for (int kc = 0; kc < D_DIM; kc += KC) {
    const int cur = (kc >> 5) & 1;

    // Current buffer's DMA complete (own wave), then barrier so all waves'
    // DMA and the previous iteration's reads of the other buffer are done.
    asm volatile("s_wait_asynccnt 0x0" ::: "memory");
    __syncthreads();

    // Overlap: issue DMA for the next chunk into the other buffer while the
    // WMMA pipe chews on this one.
    if (kc + KC < D_DIM)
      async_stage(Ah_g, Al_g, Bh_g, Bl_g, smem, cur ^ 1, tid, rowBase, colBase,
                  kc + KC);

    const __bf16* Ah = (const __bf16*)(smem + cur * BUF_BYTES + 0 * ARR_BYTES);
    const __bf16* Al = (const __bf16*)(smem + cur * BUF_BYTES + 1 * ARR_BYTES);
    const __bf16* Bh = (const __bf16*)(smem + cur * BUF_BYTES + 2 * ARR_BYTES);
    const __bf16* Bl = (const __bf16*)(smem + cur * BUF_BYTES + 3 * ARR_BYTES);

    v16bf a_h[2], a_l[2];
    #pragma unroll
    for (int i = 0; i < 2; ++i) {
      a_h[i] = load_frag(Ah + (rg * 32 + i * 16) * STRIDE, lane);
      a_l[i] = load_frag(Al + (rg * 32 + i * 16) * STRIDE, lane);
    }
    #pragma unroll
    for (int j = 0; j < 4; ++j) {
      const v16bf b_h = load_frag(Bh + (cg * 64 + j * 16) * STRIDE, lane);
      const v16bf b_l = load_frag(Bl + (cg * 64 + j * 16) * STRIDE, lane);
      #pragma unroll
      for (int i = 0; i < 2; ++i) {
        // dot += hi*hi + hi*lo + lo*hi  (f32 accumulate)
        acc[i * 4 + j] = __builtin_amdgcn_wmma_f32_16x16x32_bf16(
            false, a_h[i], false, b_h, (short)0, acc[i * 4 + j], false, false);
        acc[i * 4 + j] = __builtin_amdgcn_wmma_f32_16x16x32_bf16(
            false, a_h[i], false, b_l, (short)0, acc[i * 4 + j], false, false);
        acc[i * 4 + j] = __builtin_amdgcn_wmma_f32_16x16x32_bf16(
            false, a_l[i], false, b_h, (short)0, acc[i * 4 + j], false, false);
      }
    }
  }

  // Epilogue: C/D layout (ISA 7.12.2): lane L, VGPR r -> M = r + 8*(L/16),
  // N = L%16 within the 16x16 tile.
  const int rl   = lane & 15;
  const int half = lane >> 4;
  float local = 0.f;
  #pragma unroll
  for (int j = 0; j < 4; ++j) {
    const int col  = colBase + cg * 64 + j * 16 + rl;
    const float bn = nB[col];
    #pragma unroll
    for (int i = 0; i < 2; ++i) {
      const int row0 = rowBase + rg * 32 + i * 16 + half * 8;
      #pragma unroll
      for (int r = 0; r < 8; ++r) {
        const float d = nA[row0 + r] + bn - 2.f * acc[i * 4 + j][r];
        local += __expf(-0.5f * d);   // sigma = 1
      }
    }
  }
  local *= w / ((float)dA * (float)dB);

  float* red = (float*)(smem + 2 * BUF_BYTES);
  red[tid] = local;
  __syncthreads();
  for (int s = 128; s > 0; s >>= 1) {
    if (tid < s) red[tid] += red[tid + s];
    __syncthreads();
  }
  if (tid == 0) atomicAdd(out, red[0]);
}

extern "C" void kernel_launch(void* const* d_in, const int* in_sizes, int n_in,
                              void* d_out, int out_size, void* d_ws, size_t ws_size,
                              hipStream_t stream) {
  const float* src = (const float*)d_in[0];
  const float* tgt = (const float*)d_in[1];
  const int N = in_sizes[0] / D_DIM;   // 8192
  const int M = in_sizes[1] / D_DIM;   // 8192

  // Workspace layout: bf16 hi/lo for both matrices (+ f32 row norms).
  char* ws = (char*)d_ws;
  size_t off = 0;
  __bf16* sHi = (__bf16*)(ws + off); off += (size_t)N * D_DIM * sizeof(__bf16);
  __bf16* sLo = (__bf16*)(ws + off); off += (size_t)N * D_DIM * sizeof(__bf16);
  __bf16* tHi = (__bf16*)(ws + off); off += (size_t)M * D_DIM * sizeof(__bf16);
  __bf16* tLo = (__bf16*)(ws + off); off += (size_t)M * D_DIM * sizeof(__bf16);
  float*  sNorm = (float*)(ws + off); off += (size_t)N * sizeof(float);
  float*  tNorm = (float*)(ws + off); off += (size_t)M * sizeof(float);

  convert_rows<<<dim3((N + 7) / 8), dim3(256), 0, stream>>>(src, sHi, sLo, sNorm, N);
  convert_rows<<<dim3((M + 7) / 8), dim3(256), 0, stream>>>(tgt, tHi, tLo, tNorm, M);
  zero_out_kernel<<<1, 1, 0, stream>>>((float*)d_out);
  mmd_tile_kernel<<<dim3(N / TILE, N / TILE, 3), dim3(256), SMEM_BYTES, stream>>>(
      sHi, sLo, tHi, tLo, sNorm, tNorm, (float*)d_out, N, M);
}
